// CausalAttention_12472585027892
// MI455X (gfx1250) — matrix operand order repaired and checked by reading, not compiled, not run on hardware
//
#include <hip/hip_runtime.h>
#include <hip/hip_bf16.h>

// Problem dims (fixed by reference)
#define BB  2
#define SS  2048
#define DD  1024
#define HH  16
#define HDD 64

typedef __bf16 bf16_t;
typedef __attribute__((ext_vector_type(16))) __bf16 v16bf;
typedef __attribute__((ext_vector_type(8)))  float  v8f;

union FragAB { v16bf v; unsigned int u[8]; };
union FragC  { v8f   v; float f[8]; };
union Pack8  { bf16_t h[8]; uint4 q; };

// ---- gfx1250 async global->LDS staging (ASYNCcnt path), guarded -----------
#if __has_builtin(__builtin_amdgcn_global_load_async_to_lds_b128) && \
    __has_builtin(__builtin_amdgcn_s_wait_asynccnt)
#define HAVE_ASYNC_LDS 1
#else
#define HAVE_ASYNC_LDS 0
#warning "gfx1250 async global->LDS builtins unavailable; using synchronous staging"
#endif

#define AS1 __attribute__((address_space(1)))
#define AS3 __attribute__((address_space(3)))

typedef int v4i_t __attribute__((vector_size(16)));
typedef AS1 v4i_t* gv4_t;   // global (AS1) int4*
typedef AS3 v4i_t* lv4_t;   // LDS (AS3) int4*

// copy 16 bytes (8 bf16) from global to LDS
__device__ __forceinline__ void copy16_g2l(const bf16_t* g, bf16_t* l) {
#if HAVE_ASYNC_LDS
  __builtin_amdgcn_global_load_async_to_lds_b128(
      (gv4_t)(unsigned long long)(uintptr_t)g,
      (lv4_t)(unsigned int)(uintptr_t)l, 0, 0);
#else
  *(uint4*)l = *(const uint4*)g;
#endif
}

__device__ __forceinline__ void staging_fence() {
#if HAVE_ASYNC_LDS
  __builtin_amdgcn_s_wait_asynccnt(0);
#endif
  __syncthreads();
}

// ---------------------------------------------------------------- fp32 -> bf16
__global__ void cvt_f32_to_bf16(const float* __restrict__ x,
                                bf16_t* __restrict__ y, int n) {
  int i = blockIdx.x * blockDim.x + threadIdx.x;
  if (i < n) y[i] = (bf16_t)x[i];
}

// ---------------------------------------------------------------- QKV proj
// grid.x = B*H*(S/64), block = 128 (4 waves, 16 rows each)
// Out[b][h][s][e] = sum_d X[b][s][d] * W[h][e][d] + bias[h][e]   (bf16 out)
// vt=0: Out laid out [B,H,S,HD];  vt=1: transposed [B,H,HD,S] (for V)
__global__ void __launch_bounds__(128) qkv_proj_kernel(
    const bf16_t* __restrict__ X,     // [B,S,D]
    const bf16_t* __restrict__ W,     // [H,HD,D]
    const float*  __restrict__ bias,  // [H,HD]
    bf16_t* __restrict__ Out,
    int vt)
{
  __shared__ __align__(16) bf16_t xs[64][32];
  __shared__ __align__(16) bf16_t ws[64][32];

  const int bh   = blockIdx.x / (SS / 64);
  const int st   = blockIdx.x % (SS / 64);
  const int b    = bh / HH;
  const int h    = bh % HH;
  const int row0 = st * 64;
  const int wave = threadIdx.x >> 5;
  const int lane = threadIdx.x & 31;
  const int half = lane >> 4;
  const int lm   = lane & 15;

  const bf16_t* Xb = X + ((size_t)b * SS + row0) * DD;
  const bf16_t* Wh = W + (size_t)h * HDD * DD;

  FragC acc[4];
#pragma unroll
  for (int t = 0; t < 4; t++) acc[t].v = (v8f)(0.0f);

  const int lr = threadIdx.x >> 1;        // 0..63
  const int lc = (threadIdx.x & 1) * 16;  // 0 or 16

  for (int k0 = 0; k0 < DD; k0 += 32) {
    copy16_g2l(Xb + (size_t)lr * DD + k0 + lc,     &xs[lr][lc]);
    copy16_g2l(Xb + (size_t)lr * DD + k0 + lc + 8, &xs[lr][lc + 8]);
    copy16_g2l(Wh + (size_t)lr * DD + k0 + lc,     &ws[lr][lc]);
    copy16_g2l(Wh + (size_t)lr * DD + k0 + lc + 8, &ws[lr][lc + 8]);
    if (k0 + 32 < DD) {  // gfx1250 global_prefetch_b8 for the next k-tile
      __builtin_prefetch(Xb + (size_t)lr * DD + k0 + 32 + lc, 0, 1);
      __builtin_prefetch(Wh + (size_t)lr * DD + k0 + 32 + lc, 0, 1);
    }
    staging_fence();

    // A fragment: 16x32 bf16, rows of X (contiguous k = 16*half + s, matching B)
    FragAB a;
    const int am = wave * 16 + lm;
#pragma unroll
    for (int i = 0; i < 8; i++) {
      const int kb = 2 * i + 16 * half;
      a.u[i] = *(const unsigned int*)&xs[am][kb];
    }
#pragma unroll
    for (int t = 0; t < 4; t++) {
      // B fragment: 32x16, column n = output feature, B[k][n] = W[n][k]
      FragAB wb;
      const int n = t * 16 + lm;
#pragma unroll
      for (int i = 0; i < 8; i++) {
        const int kb = 2 * i + 16 * half;
        wb.u[i] = *(const unsigned int*)&ws[n][kb];
      }
      acc[t].v = __builtin_amdgcn_wmma_f32_16x16x32_bf16(
          false, a.v, false, wb.v, (short)0, acc[t].v, false, false);
    }
    __syncthreads();
  }

  if (vt) {
    // transposed store: Out[((b*H+h)*HD + n)*S + s]; 8 consecutive rows/lane
#pragma unroll
    for (int t = 0; t < 4; t++) {
      const int n = t * 16 + lm;
      const float bv = bias[h * HDD + n];
      Pack8 pk;
#pragma unroll
      for (int i = 0; i < 8; i++) pk.h[i] = (bf16_t)(acc[t].f[i] + bv);
      bf16_t* dst = Out + (((size_t)b * HH + h) * HDD + n) * SS
                        + row0 + wave * 16 + 8 * half;
      *(uint4*)dst = pk.q;
    }
  } else {
    bf16_t* Ob = Out + (((size_t)b * HH + h) * SS + row0 + wave * 16) * HDD;
#pragma unroll
    for (int t = 0; t < 4; t++) {
      const int n = t * 16 + lm;
      const float bv = bias[h * HDD + n];
#pragma unroll
      for (int i = 0; i < 8; i++) {
        const int m = i + 8 * half;  // C layout: VGPR i -> row i (+8 hi half)
        Ob[(size_t)m * HDD + n] = (bf16_t)(acc[t].f[i] + bv);
      }
    }
  }
}

// ---------------------------------------------------------------- attention
// grid.x = B*H*(S/64), block = 128. Flash-attention, 64-row q tile per block,
// 32-key tiles (one WMMA K-step), online softmax in fp32.
// V is pre-transposed: Vt[B,H,HD,S] so all LDS fragment reads are pair-loads.
__global__ void __launch_bounds__(128) attn_kernel(
    const bf16_t* __restrict__ Q,   // [B,H,S,HD]
    const bf16_t* __restrict__ K,   // [B,H,S,HD]
    const bf16_t* __restrict__ Vt,  // [B,H,HD,S]
    bf16_t* __restrict__ Z)         // [B,S,D] (heads concatenated)
{
  __shared__ __align__(16) bf16_t qs[64][64];
  __shared__ __align__(16) bf16_t ks[32][64];
  __shared__ __align__(16) bf16_t vst[64][32];    // [hd][key]
  __shared__ __align__(16) bf16_t ps[4][16][32];  // per-wave P tile

  const int bh  = blockIdx.x / (SS / 64);
  const int qt  = blockIdx.x % (SS / 64);
  const int b   = bh / HH;
  const int h   = bh % HH;
  const int q0  = qt * 64;
  const int wave = threadIdx.x >> 5;
  const int lane = threadIdx.x & 31;
  const int half = lane >> 4;
  const int lm   = lane & 15;

  const bf16_t* Qb  = Q  + (((size_t)b * HH + h) * SS + q0) * HDD;
  const bf16_t* Kb  = K  + (((size_t)b * HH + h) * SS) * HDD;
  const bf16_t* Vtb = Vt + (((size_t)b * HH + h) * HDD) * SS;

  // load Q tile 64x64
  {
    const int r = threadIdx.x >> 1;
    const int c = (threadIdx.x & 1) * 32;
    copy16_g2l(Qb + (size_t)r * HDD + c,      &qs[r][c]);
    copy16_g2l(Qb + (size_t)r * HDD + c + 8,  &qs[r][c + 8]);
    copy16_g2l(Qb + (size_t)r * HDD + c + 16, &qs[r][c + 16]);
    copy16_g2l(Qb + (size_t)r * HDD + c + 24, &qs[r][c + 24]);
  }
  staging_fence();

  // Q as two A fragments (HD=64 -> 2 K-steps of 32)
  FragAB qa[2];
  const int am = wave * 16 + lm;
#pragma unroll
  for (int s = 0; s < 2; s++)
#pragma unroll
    for (int i = 0; i < 8; i++) {
      const int kb = 2 * i + 16 * half + 32 * s;
      qa[s].u[i] = *(const unsigned int*)&qs[am][kb];
    }

  FragC o[4];
#pragma unroll
  for (int t = 0; t < 4; t++) o[t].v = (v8f)(0.0f);
  float row_max[8], row_sum[8];
#pragma unroll
  for (int i = 0; i < 8; i++) { row_max[i] = -3.0e38f; row_sum[i] = 0.0f; }

  const int kr = threadIdx.x >> 2;        // 0..31 (key rows of K tile)
  const int kc = (threadIdx.x & 3) * 16;  // 0,16,32,48
  const int vr = threadIdx.x >> 1;        // 0..63 (hd rows of Vt tile)
  const int vc = (threadIdx.x & 1) * 16;  // 0 or 16

  for (int j0 = 0; j0 < q0 + 64; j0 += 32) {
    __syncthreads();  // previous iteration done reading ks/vst
    copy16_g2l(Kb + (size_t)(j0 + kr) * HDD + kc,     &ks[kr][kc]);
    copy16_g2l(Kb + (size_t)(j0 + kr) * HDD + kc + 8, &ks[kr][kc + 8]);
    copy16_g2l(Vtb + (size_t)vr * SS + j0 + vc,       &vst[vr][vc]);
    copy16_g2l(Vtb + (size_t)vr * SS + j0 + vc + 8,   &vst[vr][vc + 8]);
    if (j0 + 32 < q0 + 64) {
      __builtin_prefetch(Kb + (size_t)(j0 + 32 + kr) * HDD + kc, 0, 1);
      __builtin_prefetch(Vtb + (size_t)vr * SS + j0 + 32 + vc, 0, 1);
    }
    staging_fence();

    // scores: two 16x16 f32 tiles over this 32-key window
    FragC sc[2];
#pragma unroll
    for (int nt = 0; nt < 2; nt++) {
      sc[nt].v = (v8f)(0.0f);
#pragma unroll
      for (int s = 0; s < 2; s++) {
        FragAB kf;  // B[k][n] = K[j0 + nt*16 + n][s*32 + k]
#pragma unroll
        for (int i = 0; i < 8; i++)
          kf.u[i] = *(const unsigned int*)&ks[nt * 16 + lm][2 * i + 16 * half + 32 * s];
        sc[nt].v = __builtin_amdgcn_wmma_f32_16x16x32_bf16(
            false, qa[s].v, false, kf.v, (short)0, sc[nt].v, false, false);
      }
    }

    // scale + causal mask + online softmax state update
    float nmax[8], corr[8];
#pragma unroll
    for (int i = 0; i < 8; i++) {
      const int m = q0 + wave * 16 + i + 8 * half;  // global query row
      float mx = row_max[i];
#pragma unroll
      for (int nt = 0; nt < 2; nt++) {
        const int col = j0 + nt * 16 + lm;  // global key column
        float sv = sc[nt].f[i] * 0.125f;    // 1/sqrt(64)
        sv = (col <= m) ? sv : -3.0e38f;
        sc[nt].f[i] = sv;
        mx = fmaxf(mx, sv);
      }
#pragma unroll
      for (int d = 1; d < 16; d <<= 1) mx = fmaxf(mx, __shfl_xor(mx, d, 32));
      nmax[i] = mx;
      corr[i] = __expf(row_max[i] - mx);
      row_sum[i] *= corr[i];
      row_max[i] = mx;
    }

    // P = exp(s - max): accumulate partial row sums, stage bf16 P in LDS
#pragma unroll
    for (int nt = 0; nt < 2; nt++)
#pragma unroll
      for (int i = 0; i < 8; i++) {
        const float p = __expf(sc[nt].f[i] - nmax[i]);
        row_sum[i] += p;
        ps[wave][i + 8 * half][nt * 16 + lm] = (bf16_t)p;
      }
    __syncthreads();

    // P (A-layout) x V -> accumulate into rescaled O
    FragAB pa;
#pragma unroll
    for (int i = 0; i < 8; i++) {
      const int kb = 2 * i + 16 * half;
      pa.u[i] = *(const unsigned int*)&ps[wave][lm][kb];
    }
#pragma unroll
    for (int t = 0; t < 4; t++) {
#pragma unroll
      for (int i = 0; i < 8; i++) o[t].f[i] *= corr[i];
      FragAB vb;  // B[k][n] = Vt[t*16+n][k]  (contiguous pair loads)
#pragma unroll
      for (int i = 0; i < 8; i++)
        vb.u[i] = *(const unsigned int*)&vst[t * 16 + lm][2 * i + 16 * half];
      o[t].v = __builtin_amdgcn_wmma_f32_16x16x32_bf16(
          false, pa.v, false, vb.v, (short)0, o[t].v, false, false);
    }
  }

  // finalize: reduce row sums across the 16-lane half group, divide, store
  float inv[8];
#pragma unroll
  for (int i = 0; i < 8; i++) {
    float s = row_sum[i];
#pragma unroll
    for (int d = 1; d < 16; d <<= 1) s += __shfl_xor(s, d, 32);
    inv[i] = 1.0f / s;
  }
  bf16_t* Zb = Z + ((size_t)b * SS + q0 + wave * 16) * DD + h * HDD;
#pragma unroll
  for (int t = 0; t < 4; t++) {
    const int n = t * 16 + lm;
#pragma unroll
    for (int i = 0; i < 8; i++) {
      const int m = i + 8 * half;
      Zb[(size_t)m * DD + n] = (bf16_t)(o[t].f[i] * inv[i]);
    }
  }
}

// ---------------------------------------------------------------- out proj
// grid = (B*S/64, D/64), block = 128.  out = Z @ Wo^T + bo  (fp32 out)
__global__ void __launch_bounds__(128) out_proj_kernel(
    const bf16_t* __restrict__ Zin,  // [B*S, D] bf16
    const bf16_t* __restrict__ Wo,   // [D, D]  bf16 (row-major, out = Z*Wo^T)
    const float*  __restrict__ bo,   // [D]
    float* __restrict__ out)         // [B*S, D] fp32
{
  __shared__ __align__(16) bf16_t xs[64][32];
  __shared__ __align__(16) bf16_t ws[64][32];

  const int row0 = blockIdx.x * 64;
  const int col0 = blockIdx.y * 64;
  const int wave = threadIdx.x >> 5;
  const int lane = threadIdx.x & 31;
  const int half = lane >> 4;
  const int lm   = lane & 15;

  const bf16_t* Xb = Zin + (size_t)row0 * DD;
  const bf16_t* Wb = Wo + (size_t)col0 * DD;

  FragC acc[4];
#pragma unroll
  for (int t = 0; t < 4; t++) acc[t].v = (v8f)(0.0f);

  const int lr = threadIdx.x >> 1;
  const int lc = (threadIdx.x & 1) * 16;

  for (int k0 = 0; k0 < DD; k0 += 32) {
    copy16_g2l(Xb + (size_t)lr * DD + k0 + lc,     &xs[lr][lc]);
    copy16_g2l(Xb + (size_t)lr * DD + k0 + lc + 8, &xs[lr][lc + 8]);
    copy16_g2l(Wb + (size_t)lr * DD + k0 + lc,     &ws[lr][lc]);
    copy16_g2l(Wb + (size_t)lr * DD + k0 + lc + 8, &ws[lr][lc + 8]);
    if (k0 + 32 < DD) {
      __builtin_prefetch(Xb + (size_t)lr * DD + k0 + 32 + lc, 0, 1);
      __builtin_prefetch(Wb + (size_t)lr * DD + k0 + 32 + lc, 0, 1);
    }
    staging_fence();

    FragAB a;
    const int am = wave * 16 + lm;
#pragma unroll
    for (int i = 0; i < 8; i++) {
      const int kb = 2 * i + 16 * half;
      a.u[i] = *(const unsigned int*)&xs[am][kb];
    }
#pragma unroll
    for (int t = 0; t < 4; t++) {
      FragAB wb;
      const int n = t * 16 + lm;
#pragma unroll
      for (int i = 0; i < 8; i++) {
        const int kb = 2 * i + 16 * half;
        wb.u[i] = *(const unsigned int*)&ws[n][kb];
      }
      acc[t].v = __builtin_amdgcn_wmma_f32_16x16x32_bf16(
          false, a.v, false, wb.v, (short)0, acc[t].v, false, false);
    }
    __syncthreads();
  }

  float* Ob = out + (size_t)(row0 + wave * 16) * DD + col0;
#pragma unroll
  for (int t = 0; t < 4; t++) {
    const int n = t * 16 + lm;
    const float bv = bo[col0 + n];
#pragma unroll
    for (int i = 0; i < 8; i++) {
      const int m = i + 8 * half;
      Ob[(size_t)m * DD + n] = acc[t].f[i] + bv;
    }
  }
}

// ---------------------------------------------------------------- launcher
extern "C" void kernel_launch(void* const* d_in, const int* in_sizes, int n_in,
                              void* d_out, int out_size, void* d_ws, size_t ws_size,
                              hipStream_t stream) {
  (void)in_sizes; (void)n_in; (void)out_size; (void)ws_size;

  const float* x  = (const float*)d_in[0];
  const float* Wq = (const float*)d_in[1];
  const float* bq = (const float*)d_in[2];
  const float* Wk = (const float*)d_in[3];
  const float* bk = (const float*)d_in[4];
  const float* Wv = (const float*)d_in[5];
  const float* bv = (const float*)d_in[6];
  const float* Wo = (const float*)d_in[7];
  const float* bo = (const float*)d_in[8];
  float* out = (float*)d_out;

  // workspace slices (bf16), 256B aligned
  size_t off = 0;
  auto slice = [&](size_t elems) {
    bf16_t* p = (bf16_t*)((char*)d_ws + off);
    off += ((elems * sizeof(bf16_t) + 255) / 256) * 256;
    return p;
  };
  const size_t nX = (size_t)BB * SS * DD;        // 4M
  const size_t nW = (size_t)HH * HDD * DD;       // 1M
  const size_t nP = (size_t)BB * HH * SS * HDD;  // 4M
  bf16_t* Xbf  = slice(nX);
  bf16_t* Wqbf = slice(nW);
  bf16_t* Wkbf = slice(nW);
  bf16_t* Wvbf = slice(nW);
  bf16_t* Wobf = slice((size_t)DD * DD);
  bf16_t* Qbf  = slice(nP);
  bf16_t* Kbf  = slice(nP);
  bf16_t* Vtbf = slice(nP);  // transposed V: [B,H,HD,S]
  bf16_t* Zbf  = slice(nX);

  // fp32 -> bf16 conversions
  cvt_f32_to_bf16<<<(int)((nX + 255) / 256), 256, 0, stream>>>(x, Xbf, (int)nX);
  cvt_f32_to_bf16<<<(int)((nW + 255) / 256), 256, 0, stream>>>(Wq, Wqbf, (int)nW);
  cvt_f32_to_bf16<<<(int)((nW + 255) / 256), 256, 0, stream>>>(Wk, Wkbf, (int)nW);
  cvt_f32_to_bf16<<<(int)((nW + 255) / 256), 256, 0, stream>>>(Wv, Wvbf, (int)nW);
  cvt_f32_to_bf16<<<(int)((DD * DD + 255) / 256), 256, 0, stream>>>(Wo, Wobf, DD * DD);

  // QKV projections (V stored transposed)
  const int nProjBlocks = BB * HH * (SS / 64);  // 1024
  qkv_proj_kernel<<<nProjBlocks, 128, 0, stream>>>(Xbf, Wqbf, bq, Qbf, 0);
  qkv_proj_kernel<<<nProjBlocks, 128, 0, stream>>>(Xbf, Wkbf, bk, Kbf, 0);
  qkv_proj_kernel<<<nProjBlocks, 128, 0, stream>>>(Xbf, Wvbf, bv, Vtbf, 1);

  // flash attention
  attn_kernel<<<BB * HH * (SS / 64), 128, 0, stream>>>(Qbf, Kbf, Vtbf, Zbf);

  // output projection
  dim3 gOut(BB * SS / 64, DD / 64);
  out_proj_kernel<<<gOut, 128, 0, stream>>>(Zbf, Wobf, bo, out);
}
